// SpatialEncoder_18854906430291
// MI455X (gfx1250) — matrix operand order repaired and checked by reading, not compiled, not run on hardware
//
#include <hip/hip_runtime.h>
#include <hip/hip_bf16.h>
#include <cstdint>
#include <cstddef>

// ---------------------------------------------------------------------------
// BiFormer-style bi-level routing attention for MI455X (gfx1250, wave32).
// All matmuls use v_wmma_f32_16x16x32_f16; intermediates stored f16 to halve
// HBM traffic (23.3 TB/s roof). SCALE folded into stored q. GEMM waves own
// multiple N-tiles per A fragment for A-side reuse. Attention K/V staging
// uses GLOBAL_LOAD_ASYNC_TO_LDS_B128 via inline asm (ASYNCcnt-tracked).
// ---------------------------------------------------------------------------

#define DIMC  256
#define QKC   256
#define CKV   512
#define QKVC  768
#define NB    4
#define IMG   128
#define HWIN  16
#define P2    64
#define HEADS 8
#define DH    32
#define TOPK  4
#define NPIX  (NB * IMG * IMG)   // 65536
#define SCALE 0.0625f            // 256^-0.5

// Toggle for the async global->LDS staging path (inline asm). Set to 0 to
// fall back to plain per-lane vector loads + ds_store.
#define USE_ASYNC_LDS_ASM 1

typedef __attribute__((ext_vector_type(16))) _Float16 v16h;
typedef __attribute__((ext_vector_type(8)))  _Float16 v8h;
typedef __attribute__((ext_vector_type(8)))  float    v8f;

__device__ __forceinline__ v8f wmma_f32_16x16x32(v16h a, v16h b, v8f c) {
  return __builtin_amdgcn_wmma_f32_16x16x32_f16(false, a, false, b, (short)0, c,
                                                false, false);
}

// A fragment (16x32 f16), per-lane: p -> &A[row][kchunk + (lane<16?0:8)]
__device__ __forceinline__ v16h load_a_h(const _Float16* p) {
  v8h lo = *(const v8h*)p;
  v8h hi = *(const v8h*)(p + 16);
  v16h r;
#pragma unroll
  for (int i = 0; i < 8; ++i) { r[i] = lo[i]; r[i + 8] = hi[i]; }
  return r;
}
// A fragment built from f32 source (on-the-fly f32 -> f16 convert)
__device__ __forceinline__ v16h load_a_f(const float* p) {
  v16h r;
#pragma unroll
  for (int i = 0; i < 8; ++i) {
    r[i]     = (_Float16)p[i];
    r[i + 8] = (_Float16)p[i + 16];
  }
  return r;
}
// B fragment (32x16): lane's column, K halves [0..7]+[16..23] offset by 8*half.
// p -> &Bsrc[col][kchunk + (lane<16?0:8)]  where Bsrc is [n][k] row-major
__device__ __forceinline__ v16h load_b_h(const _Float16* p) {
  v8h lo = *(const v8h*)p;
  v8h hi = *(const v8h*)(p + 16);
  v16h r;
#pragma unroll
  for (int i = 0; i < 8; ++i) { r[i] = lo[i]; r[i + 8] = hi[i]; }
  return r;
}

// ---------------------------------------------------------------------------
// Weight conversion f32 -> f16 (kept in [out][k] order so WMMA B loads are
// contiguous b128s).
// ---------------------------------------------------------------------------
__global__ void k_wcvt(const float* __restrict__ Wqkv, const float* __restrict__ Wo,
                       _Float16* __restrict__ WqkvH, _Float16* __restrict__ WoH) {
  int i = blockIdx.x * blockDim.x + threadIdx.x;
  if (i < QKVC * DIMC) WqkvH[i] = (_Float16)Wqkv[i];
  if (i < DIMC * DIMC) WoH[i] = (_Float16)Wo[i];
}

// ---------------------------------------------------------------------------
// QKV projection: [NPIX,256] @ WqkvT -> q (f16, pre-scaled by SCALE) + kv (f16)
// grid(NPIX/16), block(256): 8 waves; each wave owns 6 N-tiles (96 cols) so
// one A fragment feeds 6 WMMAs (48 WMMA per wave per tile-row).
// ---------------------------------------------------------------------------
__global__ __launch_bounds__(256) void k_qkv_gemm(
    const float* __restrict__ X, const _Float16* __restrict__ Wh,
    const float* __restrict__ bias, _Float16* __restrict__ Qo,
    _Float16* __restrict__ KVo) {
  const int lane = threadIdx.x & 31;
  const int wave = threadIdx.x >> 5;
  const int row0 = blockIdx.x * 16;
  const int ar   = lane & 15;
  const int akb  = (lane < 16) ? 0 : 8;
  const int bn   = lane & 15;
  const int bkb  = (lane < 16) ? 0 : 8;
  const int mb   = (lane < 16) ? 0 : 8;
  const int colbase = wave * 96;                    // 6 tiles of 16
  const float* arow = X + (size_t)(row0 + ar) * DIMC + akb;
  v8f acc[6] = {};
#pragma unroll
  for (int kc = 0; kc < DIMC; kc += 32) {
    v16h a = load_a_f(arow + kc);
#pragma unroll
    for (int t = 0; t < 6; ++t) {
      const _Float16* bp =
          Wh + (size_t)(colbase + t * 16 + bn) * DIMC + bkb + kc;
      acc[t] = wmma_f32_16x16x32(a, load_b_h(bp), acc[t]);
    }
  }
#pragma unroll
  for (int t = 0; t < 6; ++t) {
    const int   c  = colbase + t * 16 + bn;
    const float bv = bias[c];
#pragma unroll
    for (int j = 0; j < 8; ++j) {
      int   m = row0 + mb + j;
      float v = acc[t][j] + bv;
      if (c < QKC) Qo[(size_t)m * QKC + c] = (_Float16)(v * SCALE);
      else         KVo[(size_t)m * CKV + (c - QKC)] = (_Float16)v;
    }
  }
}

// ---------------------------------------------------------------------------
// 4x4 block-mean pooling of kv within each window -> Pool[n][win][16][512]
// ---------------------------------------------------------------------------
__global__ __launch_bounds__(256) void k_pool(const _Float16* __restrict__ KV,
                                              _Float16* __restrict__ Pool) {
  const int bid = blockIdx.x;               // ((n*64 + win)*16 + t)
  const int t = bid & 15, win = (bid >> 4) & 63, n = bid >> 10;
  const int c = threadIdx.x;
  const int by = t >> 2, bx = t & 3;
  const int wy = win >> 3, wx = win & 7;
  float s0 = 0.f, s1 = 0.f;
#pragma unroll
  for (int q = 0; q < 16; ++q) {
    int iy = by * 4 + (q >> 2), ix = bx * 4 + (q & 3);
    size_t pix = ((size_t)n * IMG + (wy * HWIN + iy)) * IMG + (wx * HWIN + ix);
    const _Float16* r = KV + pix * CKV;
    s0 += (float)r[c];
    s1 += (float)r[c + QKC];
  }
  _Float16* o = Pool + (size_t)bid * CKV;
  o[c]       = (_Float16)(s0 * (1.f / 16.f));
  o[c + QKC] = (_Float16)(s1 * (1.f / 16.f));
}

// ---------------------------------------------------------------------------
// Per-window means for routing. qw inherits the folded SCALE from stored q.
// ---------------------------------------------------------------------------
__global__ __launch_bounds__(256) void k_winstat(const _Float16* __restrict__ Q,
                                                 const _Float16* __restrict__ KV,
                                                 float* __restrict__ qw,
                                                 float* __restrict__ kw) {
  const int bid = blockIdx.x;               // n*64 + win
  const int win = bid & 63, n = bid >> 6;
  const int c = threadIdx.x;
  const int wy = win >> 3, wx = win & 7;
  float sq = 0.f, sk = 0.f;
  for (int p = 0; p < 256; ++p) {
    int iy = p >> 4, ix = p & 15;
    size_t pix = ((size_t)n * IMG + (wy * HWIN + iy)) * IMG + (wx * HWIN + ix);
    sq += (float)Q[pix * QKC + c];
    sk += (float)KV[pix * CKV + c];
  }
  qw[(size_t)bid * QKC + c] = sq * (1.f / 256.f);
  kw[(size_t)bid * QKC + c] = sk * (1.f / 256.f);
}

// ---------------------------------------------------------------------------
// Routing: logits (diag forced to 1.0), top-4 (ties -> lowest index).
// ---------------------------------------------------------------------------
__global__ void k_route(const float* __restrict__ qw, const float* __restrict__ kw,
                        int* __restrict__ idx) {
  const int bid = blockIdx.x;               // n*64 + i
  const int i = bid & 63, n = bid >> 6;
  const int j = threadIdx.x;                // 0..63
  __shared__ float lg[64];
  const float* qr = qw + (size_t)bid * QKC;
  const float* kr = kw + (size_t)(n * 64 + j) * QKC;
  float d = 0.f;
  for (int c = 0; c < QKC; ++c) d += qr[c] * kr[c];
  lg[j] = (i == j) ? 1.0f : d;
  __syncthreads();
  if (j == 0) {
    int* o = idx + (size_t)bid * TOPK;
    for (int s = 0; s < TOPK; ++s) {
      float best = -1e30f; int bi = 0;
      for (int t = 0; t < 64; ++t)
        if (lg[t] > best) { best = lg[t]; bi = t; }
      o[s] = bi;
      lg[bi] = -1e30f;
    }
  }
}

// ---------------------------------------------------------------------------
// Attention: one block per (n, window, head). K/V staged into LDS via async
// global->LDS b128 (ASYNCcnt, inline asm); V transposed inside LDS.
// Scores via WMMA (q pre-scaled) -> LDS -> row softmax -> f16 P -> P@V WMMA.
// grid(N*64, HEADS), block(256) = 8 waves over 16 M-tiles.
// ---------------------------------------------------------------------------
__global__ __launch_bounds__(256) void k_attend(
    const _Float16* __restrict__ Q, const _Float16* __restrict__ Pool,
    const int* __restrict__ Idx, _Float16* __restrict__ O) {
  const int nw = blockIdx.x;                // n*64 + win
  const int win = nw & 63, n = nw >> 6;
  const int head = blockIdx.y;
  const int tid = threadIdx.x;
  const int lane = tid & 31, wave = tid >> 5;

  __shared__ __align__(16) _Float16 kbuf[64][32];      // k_sel rows
  __shared__ __align__(16) _Float16 vbuf[64][32];      // v_sel rows (row-major)
  __shared__ __align__(16) _Float16 vtbuf[32][64];     // V^T: [c][j]
  __shared__ __align__(16) float    sbuf[8][16][64];   // per-wave scores
  __shared__ __align__(16) _Float16 pbuf[8][16][64];   // per-wave probs (f16)
  __shared__ int sel[4];

  if (tid < 4) sel[tid] = Idx[(size_t)nw * TOPK + tid];
  __syncthreads();

  {
    // 64 rows x 32 halves, 16B granules: 64*4 = 256 ops, one per thread.
    // K at +0 and V at +512B in the same pooled row share one address.
    int j = tid >> 2, c8 = (tid & 3) * 8;
    const _Float16* src =
        Pool + ((size_t)(n * 64 + sel[j >> 4]) * 16 + (j & 15)) * CKV +
        head * DH + c8;
#if USE_ASYNC_LDS_ASM
    unsigned kdst = (unsigned)(size_t)&kbuf[j][c8];
    unsigned vdst = (unsigned)(size_t)&vbuf[j][c8];
    unsigned long long gsrc = (unsigned long long)(size_t)src;
    asm volatile(
        "global_load_async_to_lds_b128 %0, %2, off\n\t"
        "global_load_async_to_lds_b128 %1, %2, off offset:512\n\t"
        "s_wait_asynccnt 0"
        :
        : "v"(kdst), "v"(vdst), "v"(gsrc)
        : "memory");
#else
    *(v8h*)&kbuf[j][c8] = *(const v8h*)src;
    *(v8h*)&vbuf[j][c8] = *(const v8h*)(src + QKC);
#endif
  }
  __syncthreads();
  for (int e = tid; e < 64 * 32; e += 256) {
    int j = e >> 5, c = e & 31;
    vtbuf[c][j] = vbuf[j][c];
  }
  __syncthreads();

  const int ar  = lane & 15;
  const int akb = (lane < 16) ? 0 : 8;
  const int bn  = lane & 15;
  const int bkb = (lane < 16) ? 0 : 8;
  const int mb  = (lane < 16) ? 0 : 8;
  const int wy = win >> 3, wx = win & 7;

  for (int mt = wave; mt < 16; mt += 8) {   // uniform: 2 iters per wave
    // scores S = q @ k^T  (16x64)
    int p = mt * 16 + ar;
    size_t pix =
        ((size_t)n * IMG + (wy * HWIN + (p >> 4))) * IMG + (wx * HWIN + (p & 15));
    v16h a = load_a_h(Q + pix * QKC + head * DH + akb);
#pragma unroll
    for (int nt = 0; nt < 4; ++nt) {
      v16h b = load_b_h(&kbuf[nt * 16 + bn][bkb]);
      v8f s = {};
      s = wmma_f32_16x16x32(a, b, s);
#pragma unroll
      for (int j = 0; j < 8; ++j) sbuf[wave][mb + j][nt * 16 + bn] = s[j];
    }
    __syncthreads();
    // row softmax over 64 keys: lanes 0..15, one row each
    if (lane < 16) {
      float mx = -1e30f;
      for (int t = 0; t < 64; ++t) mx = fmaxf(mx, sbuf[wave][lane][t]);
      float sum = 0.f;
      for (int t = 0; t < 64; ++t) {
        float e = __expf(sbuf[wave][lane][t] - mx);
        sbuf[wave][lane][t] = e;
        sum += e;
      }
      float inv = 1.f / sum;
      for (int t = 0; t < 64; ++t)
        pbuf[wave][lane][t] = (_Float16)(sbuf[wave][lane][t] * inv);
    }
    __syncthreads();
    // O = P @ V  (16x32), K = 64 -> two chunks
#pragma unroll
    for (int nt = 0; nt < 2; ++nt) {
      v8f o = {};
#pragma unroll
      for (int kc = 0; kc < 2; ++kc) {
        v16h pa = load_a_h(&pbuf[wave][ar][kc * 32 + akb]);
        v16h vb = load_b_h(&vtbuf[nt * 16 + bn][kc * 32 + bkb]);
        o = wmma_f32_16x16x32(pa, vb, o);
      }
#pragma unroll
      for (int j = 0; j < 8; ++j) {
        int pm = mt * 16 + mb + j;
        size_t opix = ((size_t)n * IMG + (wy * HWIN + (pm >> 4))) * IMG +
                      (wx * HWIN + (pm & 15));
        O[opix * DIMC + head * DH + nt * 16 + bn] = (_Float16)o[j];
      }
    }
    __syncthreads();
  }
}

// ---------------------------------------------------------------------------
// LEPE: depthwise 3x3 over full-resolution v (+bias), added in place to the
// attention output (pre-projection).
// ---------------------------------------------------------------------------
__global__ __launch_bounds__(256) void k_lepe(const _Float16* __restrict__ KV,
                                              const float* __restrict__ w,
                                              const float* __restrict__ b,
                                              _Float16* __restrict__ O) {
  const int pix = blockIdx.x;
  const int c = threadIdx.x;
  const int n = pix >> 14;
  const int y = (pix >> 7) & 127;
  const int x = pix & 127;
  float acc = b[c];
#pragma unroll
  for (int dy = 0; dy < 3; ++dy) {
    int yy = y + dy - 1;
    if (yy < 0 || yy > IMG - 1) continue;
#pragma unroll
    for (int dx = 0; dx < 3; ++dx) {
      int xx = x + dx - 1;
      if (xx < 0 || xx > IMG - 1) continue;
      size_t p = ((size_t)n * IMG + yy) * IMG + xx;
      acc += (float)KV[p * CKV + QKC + c] * w[c * 9 + dy * 3 + dx];
    }
  }
  size_t o = (size_t)pix * DIMC + c;
  O[o] = (_Float16)((float)O[o] + acc);
}

// ---------------------------------------------------------------------------
// Output projection: f16 [NPIX,256] @ Wo^T + bo -> f32 out.
// grid(NPIX/16), block(256): 8 waves, 2 N-tiles per wave (A reuse x2).
// ---------------------------------------------------------------------------
__global__ __launch_bounds__(256) void k_out_gemm(
    const _Float16* __restrict__ A, const _Float16* __restrict__ Wh,
    const float* __restrict__ bias, float* __restrict__ Out) {
  const int lane = threadIdx.x & 31;
  const int wave = threadIdx.x >> 5;
  const int row0 = blockIdx.x * 16;
  const int ar   = lane & 15;
  const int akb  = (lane < 16) ? 0 : 8;
  const int bn   = lane & 15;
  const int bkb  = (lane < 16) ? 0 : 8;
  const int mb   = (lane < 16) ? 0 : 8;
  const int colbase = wave * 32;                    // 2 tiles of 16
  const _Float16* arow = A + (size_t)(row0 + ar) * DIMC + akb;
  v8f acc[2] = {};
#pragma unroll
  for (int kc = 0; kc < DIMC; kc += 32) {
    v16h a = load_a_h(arow + kc);
#pragma unroll
    for (int t = 0; t < 2; ++t) {
      const _Float16* bp =
          Wh + (size_t)(colbase + t * 16 + bn) * DIMC + bkb + kc;
      acc[t] = wmma_f32_16x16x32(a, load_b_h(bp), acc[t]);
    }
  }
#pragma unroll
  for (int t = 0; t < 2; ++t) {
    const int   c  = colbase + t * 16 + bn;
    const float bv = bias[c];
#pragma unroll
    for (int j = 0; j < 8; ++j) {
      int m = row0 + mb + j;
      Out[(size_t)m * DIMC + c] = acc[t][j] + bv;
    }
  }
}

// ---------------------------------------------------------------------------
extern "C" void kernel_launch(void* const* d_in, const int* in_sizes, int n_in,
                              void* d_out, int out_size, void* d_ws,
                              size_t ws_size, hipStream_t stream) {
  (void)in_sizes; (void)n_in; (void)out_size; (void)ws_size;
  const float* x[2]    = {(const float*)d_in[0], (const float*)d_in[1]};
  const float* Wqkv    = (const float*)d_in[2];
  const float* bqkv    = (const float*)d_in[3];
  const float* lepe_w  = (const float*)d_in[4];
  const float* lepe_b  = (const float*)d_in[5];
  const float* Wo      = (const float*)d_in[6];
  const float* bo      = (const float*)d_in[7];

  char* ws = (char*)d_ws;
  size_t off = 0;
  auto bump = [&](size_t bytes) -> char* {
    char* p = ws + off;
    off = (off + bytes + 255) & ~(size_t)255;
    return p;
  };
  _Float16* WqkvH = (_Float16*)bump((size_t)QKVC * DIMC * 2);
  _Float16* WoH   = (_Float16*)bump((size_t)DIMC * DIMC * 2);
  _Float16* Qb[2], *KVb[2], *Poolb[2], *Attnb[2];
  float *QWb[2], *KWb[2];
  int* Idxb[2];
  for (int b = 0; b < 2; ++b) Qb[b]    = (_Float16*)bump((size_t)NPIX * QKC * 2);
  for (int b = 0; b < 2; ++b) KVb[b]   = (_Float16*)bump((size_t)NPIX * CKV * 2);
  for (int b = 0; b < 2; ++b) Poolb[b] = (_Float16*)bump((size_t)NB * P2 * 16 * CKV * 2);
  for (int b = 0; b < 2; ++b) QWb[b]   = (float*)bump((size_t)NB * P2 * QKC * 4);
  for (int b = 0; b < 2; ++b) KWb[b]   = (float*)bump((size_t)NB * P2 * QKC * 4);
  for (int b = 0; b < 2; ++b) Idxb[b]  = (int*)bump((size_t)NB * P2 * TOPK * 4);
  for (int b = 0; b < 2; ++b) Attnb[b] = (_Float16*)bump((size_t)NPIX * DIMC * 2);

  float* out1 = (float*)d_out;
  float* out2 = out1 + (size_t)NPIX * DIMC;

  k_wcvt<<<(QKVC * DIMC + 255) / 256, 256, 0, stream>>>(Wqkv, Wo, WqkvH, WoH);

  for (int b = 0; b < 2; ++b)
    k_qkv_gemm<<<NPIX / 16, 256, 0, stream>>>(x[b], WqkvH, bqkv, Qb[b], KVb[b]);

  for (int b = 0; b < 2; ++b)
    k_pool<<<NB * P2 * 16, 256, 0, stream>>>(KVb[b], Poolb[b]);

  for (int b = 0; b < 2; ++b)
    k_winstat<<<NB * P2, 256, 0, stream>>>(Qb[b], KVb[b], QWb[b], KWb[b]);

  // idx1 = route(q2_win, k1_win); idx2 = route(q1_win, k2_win)
  k_route<<<NB * P2, 64, 0, stream>>>(QWb[1], KWb[0], Idxb[0]);
  k_route<<<NB * P2, 64, 0, stream>>>(QWb[0], KWb[1], Idxb[1]);

  // out1 = attend(q2, kv1_pool, idx1) + lepe(v1); out2 likewise swapped
  k_attend<<<dim3(NB * P2, HEADS), 256, 0, stream>>>(Qb[1], Poolb[0], Idxb[0], Attnb[0]);
  k_attend<<<dim3(NB * P2, HEADS), 256, 0, stream>>>(Qb[0], Poolb[1], Idxb[1], Attnb[1]);

  k_lepe<<<NPIX, 256, 0, stream>>>(KVb[0], lepe_w, lepe_b, Attnb[0]);
  k_lepe<<<NPIX, 256, 0, stream>>>(KVb[1], lepe_w, lepe_b, Attnb[1]);

  k_out_gemm<<<NPIX / 16, 256, 0, stream>>>(Attnb[0], WoH, bo, out1);
  k_out_gemm<<<NPIX / 16, 256, 0, stream>>>(Attnb[1], WoH, bo, out2);
}
